// QuantumImagePreprocessor_51307679318825
// MI455X (gfx1250) — matrix operand order repaired and checked by reading, not compiled, not run on hardware
//
#include <hip/hip_runtime.h>

typedef __attribute__((ext_vector_type(16))) _Float16 v16h;
typedef __attribute__((ext_vector_type(8)))  float    v8f;

struct cplx { float re, im; };
__device__ __forceinline__ cplx cmul(cplx a, cplx b) {
  return { a.re*b.re - a.im*b.im, a.re*b.im + a.im*b.re };
}

// Image: 32 x 224 x 224 ; patches: 32 x 112 x 112 (16 per tile)
// tiles per image row = 112/16 = 7 ; tiles per image = 112*7 = 784 ; total tiles = 25088

// ---------------------------------------------------------------------------
// Setup: simulate the 2-layer StronglyEntanglingLayers circuit on all 16 basis
// states to build U (16x16 complex), then emit the two real-ified 32x16 f16
// B-matrices in the exact WMMA per-lane layout.
// ---------------------------------------------------------------------------
__global__ void q_build_unitary_51307679318825(const float* __restrict__ wts,
                                               _Float16* __restrict__ Bbuf) {
  if (threadIdx.x != 0 || blockIdx.x != 0) return;
  cplx U[16][16];
  for (int b = 0; b < 16; ++b) {
    cplx a[16];
    for (int k = 0; k < 16; ++k) a[k] = {0.f, 0.f};
    a[b] = {1.f, 0.f};
    for (int l = 0; l < 2; ++l) {
      for (int q = 0; q < 4; ++q) {
        float phi = wts[(l*4+q)*3+0], th = wts[(l*4+q)*3+1], om = wts[(l*4+q)*3+2];
        float c  = __cosf(0.5f*th), s = __sinf(0.5f*th);
        float ap = 0.5f*(phi+om),  am = 0.5f*(phi-om);
        cplx ep = { __cosf(ap), -__sinf(ap) };   // e^{-i(phi+om)/2}
        cplx em = { __cosf(am), -__sinf(am) };   // e^{-i(phi-om)/2}
        cplx m00 = {  ep.re*c,  ep.im*c };       // ep*c
        cplx m01 = { -em.re*s,  em.im*s };       // -conj(em)*s
        cplx m10 = {  em.re*s,  em.im*s };       // em*s
        cplx m11 = {  ep.re*c, -ep.im*c };       // conj(ep)*c
        int mask = 1 << (3 - q);                 // wire 0 = MSB
        for (int k = 0; k < 16; ++k) if (!(k & mask)) {
          cplx a0 = a[k], a1 = a[k|mask];
          cplx n0 = { m00.re*a0.re - m00.im*a0.im + m01.re*a1.re - m01.im*a1.im,
                      m00.re*a0.im + m00.im*a0.re + m01.re*a1.im + m01.im*a1.re };
          cplx n1 = { m10.re*a0.re - m10.im*a0.im + m11.re*a1.re - m11.im*a1.im,
                      m10.re*a0.im + m10.im*a0.re + m11.re*a1.im + m11.im*a1.re };
          a[k] = n0; a[k|mask] = n1;
        }
      }
      int r = (l % 3) + 1;
      for (int q = 0; q < 4; ++q) {
        int tq = (q + r) & 3;
        int cm = 1 << (3 - q), tm = 1 << (3 - tq);
        for (int k = 0; k < 16; ++k) if ((k & cm) && !(k & tm)) {
          cplx tmp = a[k]; a[k] = a[k|tm]; a[k|tm] = tmp;
        }
      }
    }
    for (int j = 0; j < 16; ++j) U[j][b] = a[j];
  }
  // B layout: element e of lane's v16h -> (K = e<8 ? 8*(lane/16)+e : 16+8*(lane/16)+(e-8), N = lane%16)
  // A row = [re(psi_in)(K=0..15), im(psi_in)(K=16..31)]
  //   B_re[K][n] = K<16 ?  Re U[n][K] : -Im U[n][K-16]
  //   B_im[K][n] = K<16 ?  Im U[n][K] :  Re U[n][K-16]
  for (int lane = 0; lane < 32; ++lane) {
    int hh = lane >> 4, n = lane & 15;
    for (int e = 0; e < 16; ++e) {
      float bre, bim;
      int K = (e < 8) ? 8*hh + e : 16 + 8*hh + (e - 8);
      if (K < 16) { bre =  U[n][K].re; bim = U[n][K].im; }
      else        { bre = -U[n][K-16].im; bim = U[n][K-16].re; }
      Bbuf[lane*16 + e]       = (_Float16)bre;
      Bbuf[512 + lane*16 + e] = (_Float16)bim;
    }
  }
}

// One Walsh-Hadamard butterfly stage across the 16-lane halves using
// ds_swizzle with an immediate group-of-32 XOR control (no index VGPRs),
// sign folded into a single FMA per value.
template <int MASK>
__device__ __forceinline__ void wht_stage_51307679318825(float (&pr)[8], int lane) {
  const float sgn = (lane & MASK) ? -1.0f : 1.0f;
  constexpr int ctl = 0x1f | (MASK << 10);   // and=0x1f, or=0, xor=MASK
#pragma unroll
  for (int v = 0; v < 8; ++v) {
    int o = __builtin_amdgcn_ds_swizzle(__builtin_bit_cast(int, pr[v]), ctl);
    pr[v] = __builtin_fmaf(pr[v], sgn, __builtin_bit_cast(float, o));
  }
}

// ---------------------------------------------------------------------------
// Main: one wave = 16 patches. Build the f16 A-matrix (product state, re|im),
// two WMMAs against B_re / B_im, square-sum, 4-stage cross-lane WHT butterfly
// for the signed Z reductions, scatter 4 outputs per patch.
// ---------------------------------------------------------------------------
__global__ __launch_bounds__(256)
void q_patch_kernel_51307679318825(const float* __restrict__ x,
                                   const _Float16* __restrict__ Bbuf,
                                   float* __restrict__ out) {
  const int lane = threadIdx.x & 31;
  const int tile = blockIdx.x * 8 + (threadIdx.x >> 5);   // 3136*8 = 25088 exact
  const int h    = lane >> 4;
  const int m    = lane & 15;

  const v16h b_re = *(const v16h*)(Bbuf + lane*16);
  const v16h b_im = *(const v16h*)(Bbuf + 512 + lane*16);

  // tile -> (image, patch-row, patch-col-base); lane m -> this lane's patch
  int bimg = tile / 784;
  int r0   = tile - bimg*784;
  int irow = r0 / 7;
  int jt   = r0 - irow*7;
  int jcol = jt*16 + m;

  const float* p = x + ((bimg*224 + 2*irow)*224 + 2*jcol);
  float2 top = *(const float2*)p;
  float2 bot = *(const float2*)(p + 224);
  float vals[4] = { top.x, top.y, bot.x, bot.y };

  // per-wire state vectors:  v0 = c^2 + i s^2 ; v1 = cs - i cs   (half = x*pi/2)
  cplx v0[4], v1[4];
#pragma unroll
  for (int q = 0; q < 4; ++q) {
    float hang = vals[q] * 1.57079632679489662f;
    float c = __cosf(hang), s = __sinf(hang);
    v0[q] = { c*c,  s*s };
    v1[q] = { c*s, -c*s };
  }

  // 8 amplitudes per lane; lane half selects wire-0 component (state MSB)
  cplx w0 = h ? v1[0] : v0[0];
  cplx t0 = cmul(w0, v0[1]);
  cplx t1 = cmul(w0, v1[1]);
  cplx u[4];
  u[0] = cmul(t0, v0[2]); u[1] = cmul(t0, v1[2]);
  u[2] = cmul(t1, v0[2]); u[3] = cmul(t1, v1[2]);

  v16h a;
#pragma unroll
  for (int jj = 0; jj < 8; ++jj) {
    cplx amp = cmul(u[jj >> 1], (jj & 1) ? v1[3] : v0[3]);
    a[jj]     = (_Float16)amp.re;   // K = 0..7 (+8 per half)  : re
    a[jj + 8] = (_Float16)amp.im;   // K = 16..23 (+8 per half): im
  }

  v8f cz = {0.f,0.f,0.f,0.f,0.f,0.f,0.f,0.f};
  v8f dre = __builtin_amdgcn_wmma_f32_16x16x32_f16(false, a, false, b_re, (short)0, cz, false, false);
  v8f dim = __builtin_amdgcn_wmma_f32_16x16x32_f16(false, a, false, b_im, (short)0, cz, false, false);

  // probabilities, then Walsh-Hadamard butterfly across the 16-lane halves:
  // lane n ends with sum_j p[j] * (-1)^{popcount(j & n)}; Z expvals sit at n=8,4,2,1
  float pr[8];
#pragma unroll
  for (int v = 0; v < 8; ++v) pr[v] = dre[v]*dre[v] + dim[v]*dim[v];

  wht_stage_51307679318825<1>(pr, lane);
  wht_stage_51307679318825<2>(pr, lane);
  wht_stage_51307679318825<4>(pr, lane);
  wht_stage_51307679318825<8>(pr, lane);

  int  wsel   = (m == 8) ? 0 : (m == 4) ? 1 : (m == 2) ? 2 : 3;
  bool active = (m == 1) || (m == 2) || (m == 4) || (m == 8);
  if (active) {
    long pb = (long)tile*16 + h*8;   // D row M = v + 8*h -> patch tile*16+M
#pragma unroll
    for (int v = 0; v < 8; ++v) out[(pb + v)*4 + wsel] = pr[v];
  }
}

extern "C" void kernel_launch(void* const* d_in, const int* in_sizes, int n_in,
                              void* d_out, int out_size, void* d_ws, size_t ws_size,
                              hipStream_t stream) {
  const float* x   = (const float*)d_in[0];   // (32,224,224,1) f32
  const float* wts = (const float*)d_in[1];   // (2,4,3) f32
  float* out       = (float*)d_out;           // (32,112,112,4) f32
  _Float16* Bbuf   = (_Float16*)d_ws;         // 2 * 32 * 16 f16 = 2 KB

  q_build_unitary_51307679318825<<<1, 1, 0, stream>>>(wts, Bbuf);
  q_patch_kernel_51307679318825<<<3136, 256, 0, stream>>>(x, Bbuf, out);
}
